// BaselineGNNRegressor_70454643523904
// MI455X (gfx1250) — matrix operand order repaired and checked by reading, not compiled, not run on hardware
//
#include <hip/hip_runtime.h>

#define N_NODES 30000
#define N_EDGES 480000
#define NGRAPH  256
#define HID     128
#define NF      64
#define EF      16
#define INFINAL (2 * HID + 4)   // 260
#define BN_EPS  1e-5f

typedef __attribute__((ext_vector_type(16))) _Float16 v16h;
typedef __attribute__((ext_vector_type(8)))  float    v8f;
typedef __attribute__((ext_vector_type(4)))  float    v4f;

__device__ __forceinline__ int laneid() { return (int)(threadIdx.x & 31); }

// ---------------------------------------------------------------------------
// WMMA fragment loaders (CDNA5 16x16x32 f16 layouts, ISA 7.12.2)
// A 16x32 (MxK): lane L holds row M=L&15; elem i -> k = {v<4: 2v+s, else 16+2(v-4)+s} + 8*(L>=16)
// B 32x16 (KxN): lane L holds col N=L&15; elem i -> k = {i<8: i, else 16+(i-8)} + 8*(L>=16)
// C 16x16:       VGPR r: lanes0-15 M=r, lanes16-31 M=r+8; N = L&15
//
// Guarded loaders are BRANCHLESS: clamp address (always in-bounds), load
// unconditionally, select zero on the value -> v_cndmask, no exec masking,
// no per-element s_wait_loadcnt.
// ---------------------------------------------------------------------------
__device__ __forceinline__ v16h load_a_frag_guard(const float* __restrict__ src, int ld,
                                                  int m0, int k0, int kmax) {
  const int lane = laneid();
  const int h = lane >> 4;
  const float* row = src + (m0 + (lane & 15)) * ld;
  v16h a;
#pragma unroll
  for (int i = 0; i < 16; ++i) {
    const int v = i >> 1, s = i & 1;
    const int kb = (v < 4) ? (2 * v) : (16 + 2 * (v - 4));
    const int k = k0 + kb + s + 8 * h;
    const int kc = (k < kmax) ? k : (kmax - 1);   // clamp: address always valid
    float f = row[kc];                            // unconditional load
    f = (k < kmax) ? f : 0.0f;                    // branchless zero-fill
    a[i] = (_Float16)f;
  }
  return a;
}

// A fragment for K=16 (zero-padded to 32): elems 0..7 = row[8h + 0..7], elems 8..15 = 0.
__device__ __forceinline__ v16h load_a_frag_k16(const float* __restrict__ src, int ld, int m0) {
  const int lane = laneid();
  const int h = lane >> 4;
  const float* er = src + (m0 + (lane & 15)) * ld + 8 * h;
  const v4f p0 = *reinterpret_cast<const v4f*>(er);
  const v4f p1 = *reinterpret_cast<const v4f*>(er + 4);
  v16h a;
#pragma unroll
  for (int i = 0; i < 4; ++i) {
    a[i]     = (_Float16)p0[i];
    a[i + 4] = (_Float16)p1[i];
  }
#pragma unroll
  for (int i = 8; i < 16; ++i) a[i] = (_Float16)0.0f;
  return a;
}

// A fragment from a 16xHID row-major LDS tile (always in-bounds)
__device__ __forceinline__ v16h load_a_frag_lds(const float* src, int k0) {
  const int lane = laneid();
  const int h = lane >> 4;
  const float* row = src + (lane & 15) * HID;
  v16h a;
#pragma unroll
  for (int i = 0; i < 16; ++i) {
    const int v = i >> 1, s = i & 1;
    const int kb = (v < 4) ? (2 * v) : (16 + 2 * (v - 4));
    a[i] = (_Float16)row[k0 + kb + s + 8 * h];
  }
  return a;
}

// Unguarded B: caller guarantees k0+31 < rows(W)
__device__ __forceinline__ v16h load_b_frag(const float* __restrict__ W, int ldw,
                                            int k0, int n0) {
  const int lane = laneid();
  const int n = n0 + (lane & 15);
  const int h = lane >> 4;
  const float* col = W + n + (k0 + 8 * h) * ldw;
  v16h b;
#pragma unroll
  for (int i = 0; i < 16; ++i) b[i] = (_Float16)col[((i < 8) ? i : i + 8) * ldw];
  return b;
}

// Guarded B (branchless clamp+select) for K not a multiple of 32
__device__ __forceinline__ v16h load_b_frag_guard(const float* __restrict__ W, int ldw,
                                                  int k0, int n0, int kmax) {
  const int lane = laneid();
  const int n = n0 + (lane & 15);
  const int h = lane >> 4;
  v16h b;
#pragma unroll
  for (int i = 0; i < 16; ++i) {
    const int k = k0 + 8 * h + ((i < 8) ? i : i + 8);
    const int kc = (k < kmax) ? k : (kmax - 1);
    float f = W[kc * ldw + n];
    f = (k < kmax) ? f : 0.0f;
    b[i] = (_Float16)f;
  }
  return b;
}

// ---------------------------------------------------------------------------
// Fused GINE edge pass: msg = relu(x[src] + edge_attr@We + be); agg[dst] += msg
// Phase 1: one WMMA per 16-col tile (fully unrolled), result (+be) -> LDS.
// Phase 2: lane = (row, half); b128 loads of x[src], ds_load_b128 of msg,
//          relu, 4 independent global_atomic_add_f32 per step.
// ---------------------------------------------------------------------------
template <int IND>
__global__ void __launch_bounds__(128)
edge_gine_kernel(const float* __restrict__ x, int ldx,
                 const float* __restrict__ eattr,
                 const int* __restrict__ eidx,      // [0,E)=src, [E,2E)=dst
                 const float* __restrict__ We,      // EF x IND
                 const float* __restrict__ be,      // IND
                 float* __restrict__ agg) {         // N x HID (ld=HID)
  __shared__ float msgT[4][16 * HID];
  const int wave = threadIdx.x >> 5;
  float* msg = &msgT[wave][0];
  const int tile = blockIdx.x * 4 + wave;
  const int e0 = tile * 16;
  const int lane = laneid();
  const int h = lane >> 4;
  const int n = lane & 15;

  // stream-ahead hint for upcoming tiles' edge features (global_prefetch_b8)
  __builtin_prefetch(&eattr[(e0 + 64) * EF], 0, 1);

  const v16h a = load_a_frag_k16(eattr, EF, e0);

#pragma unroll
  for (int nt = 0; nt < IND / 16; ++nt) {
    const v16h b = load_b_frag_guard(We, IND, 0, nt * 16, EF);
    v8f c = {};
    c = __builtin_amdgcn_wmma_f32_16x16x32_f16(false, a, false, b, (short)0, c, false, false);
    const int col = nt * 16 + n;
    const float bias = be[col];
#pragma unroll
    for (int r = 0; r < 8; ++r)
      msg[(r + 8 * h) * HID + col] = c[r] + bias;
  }

  // Phase 2: gather + relu + scatter, vectorized
  const int m = n;                       // row of this lane
  const int src = eidx[e0 + m];
  const int dst = eidx[N_EDGES + e0 + m];
  const float* xrow = x + src * ldx;
  const float* mrow = msg + m * HID;
  float* arow = agg + dst * HID;
  const int cb = h * (IND / 2);
#pragma unroll
  for (int j = 0; j < IND / 8; ++j) {
    const int c = cb + j * 4;
    const v4f xv = *reinterpret_cast<const v4f*>(xrow + c);
    const v4f mv = *reinterpret_cast<const v4f*>(mrow + c);
    const v4f sum = xv + mv;
    atomicAdd(arow + c + 0, fmaxf(sum[0], 0.0f));
    atomicAdd(arow + c + 1, fmaxf(sum[1], 0.0f));
    atomicAdd(arow + c + 2, fmaxf(sum[2], 0.0f));
    atomicAdd(arow + c + 3, fmaxf(sum[3], 0.0f));
  }
}

// ---------------------------------------------------------------------------
// Node MLP: h = x+agg; h1 = relu(h@W1+b1); h2 = h1@W2+b2; out = relu(BN(h2))
// hin staged into LDS with b128 loads; both GEMMs read A-fragments from LDS.
// One wave per 16-node tile -> in-wave DS ordering, no barriers needed.
// ---------------------------------------------------------------------------
template <int IND>
__global__ void __launch_bounds__(128)
node_mlp_kernel(const float* __restrict__ x, int ldx,
                const float* __restrict__ agg,
                const float* __restrict__ W1, const float* __restrict__ b1,
                const float* __restrict__ W2, const float* __restrict__ b2,
                const float* __restrict__ gamma, const float* __restrict__ beta,
                const float* __restrict__ mean, const float* __restrict__ var,
                float* __restrict__ out, int ntile_total) {
  __shared__ float ldsT[4][16 * HID];
  const int wave = threadIdx.x >> 5;
  const int tile = blockIdx.x * 4 + wave;
  if (tile >= ntile_total) return;
  float* tl = &ldsT[wave][0];
  const int m0 = tile * 16;
  const int lane = laneid();
  const int h = lane >> 4;
  const int n = lane & 15;

  // Stage hin = x + agg into LDS (vectorized, coalesced)
  {
    const float* xr = x + (m0 + n) * ldx;
    const float* ar = agg + (m0 + n) * HID;
    const int cb = h * (IND / 2);
#pragma unroll
    for (int j = 0; j < IND / 8; ++j) {
      const int c = cb + j * 4;
      const v4f xv = *reinterpret_cast<const v4f*>(xr + c);
      const v4f av = *reinterpret_cast<const v4f*>(ar + c);
      const v4f s = xv + av;
      *reinterpret_cast<v4f*>(tl + n * HID + c) = s;
    }
  }

  v8f acc[8];
#pragma unroll
  for (int nt = 0; nt < 8; ++nt)
#pragma unroll
    for (int r = 0; r < 8; ++r) acc[nt][r] = 0.0f;

#pragma unroll
  for (int kt = 0; kt < IND / 32; ++kt) {
    const v16h a = load_a_frag_lds(tl, kt * 32);
#pragma unroll
    for (int nt = 0; nt < 8; ++nt) {
      const v16h b = load_b_frag(W1, HID, kt * 32, nt * 16);
      acc[nt] = __builtin_amdgcn_wmma_f32_16x16x32_f16(false, a, false, b, (short)0, acc[nt], false, false);
    }
  }

  // h1 = relu(acc + b1) -> LDS (overwrites hin); reset accumulators
#pragma unroll
  for (int nt = 0; nt < 8; ++nt) {
    const int col = nt * 16 + n;
    const float bias = b1[col];
#pragma unroll
    for (int r = 0; r < 8; ++r) {
      const int m = r + 8 * h;
      tl[m * HID + col] = fmaxf(acc[nt][r] + bias, 0.0f);
      acc[nt][r] = 0.0f;
    }
  }

#pragma unroll
  for (int kt = 0; kt < HID / 32; ++kt) {
    const v16h a = load_a_frag_lds(tl, kt * 32);
#pragma unroll
    for (int nt = 0; nt < 8; ++nt) {
      const v16h b = load_b_frag(W2, HID, kt * 32, nt * 16);
      acc[nt] = __builtin_amdgcn_wmma_f32_16x16x32_f16(false, a, false, b, (short)0, acc[nt], false, false);
    }
  }

  // BN (eval) + ReLU, store
#pragma unroll
  for (int nt = 0; nt < 8; ++nt) {
    const int col = nt * 16 + n;
    const float ga = gamma[col], bt = beta[col], mu = mean[col];
    const float inv = rsqrtf(var[col] + BN_EPS);
    const float bias = b2[col];
#pragma unroll
    for (int r = 0; r < 8; ++r) {
      const int m = r + 8 * h;
      const float y = ga * (acc[nt][r] + bias - mu) * inv + bt;
      out[(m0 + m) * HID + col] = fmaxf(y, 0.0f);
    }
  }
}

// global_add_pool: b128 load + 4 atomics per thread (g is L2-resident)
__global__ void __launch_bounds__(256)
pool_kernel(const float* __restrict__ xf, const int* __restrict__ batch,
            float* __restrict__ g) {
  const int idx = blockIdx.x * 256 + (int)threadIdx.x;   // N_NODES*32 total
  const int node = idx >> 5;
  const int c = (idx & 31) << 2;
  if (node >= N_NODES) return;
  const v4f v = *reinterpret_cast<const v4f*>(xf + node * HID + c);
  float* gr = g + batch[node] * HID + c;
  atomicAdd(gr + 0, v[0]);
  atomicAdd(gr + 1, v[1]);
  atomicAdd(gr + 2, v[2]);
  atomicAdd(gr + 3, v[3]);
}

__global__ void __launch_bounds__(256)
concat_kernel(const float* __restrict__ gsv, const float* __restrict__ gsu,
              const float* __restrict__ gfeat, float* __restrict__ gc) {
  const int idx = blockIdx.x * 256 + (int)threadIdx.x;
  if (idx >= NGRAPH * INFINAL) return;
  const int row = idx / INFINAL;
  const int c = idx - row * INFINAL;
  float v;
  if (c < HID)            v = gsv[row * HID + c];
  else if (c < 2 * HID)   v = gsu[row * HID + (c - HID)];
  else                    v = gfeat[row * 4 + (c - 2 * HID)];
  gc[idx] = v;
}

// Head GEMM: g_final = relu(g_concat @ Wf + bf); K=260 zero-padded to 288
__global__ void __launch_bounds__(128)
head_mlp_kernel(const float* __restrict__ gc, const float* __restrict__ Wf,
                const float* __restrict__ bf, float* __restrict__ gfin) {
  const int wave = threadIdx.x >> 5;
  const int tile = blockIdx.x * 4 + wave;   // 16 tiles of 16 graphs
  const int m0 = tile * 16;
  const int lane = laneid();
  const int h = lane >> 4;
  const int n = lane & 15;

  v8f acc[8];
#pragma unroll
  for (int nt = 0; nt < 8; ++nt)
#pragma unroll
    for (int r = 0; r < 8; ++r) acc[nt][r] = 0.0f;

#pragma unroll
  for (int kt = 0; kt < 9; ++kt) {
    const v16h a = load_a_frag_guard(gc, INFINAL, m0, kt * 32, INFINAL);
#pragma unroll
    for (int nt = 0; nt < 8; ++nt) {
      const v16h b = load_b_frag_guard(Wf, HID, kt * 32, nt * 16, INFINAL);
      acc[nt] = __builtin_amdgcn_wmma_f32_16x16x32_f16(false, a, false, b, (short)0, acc[nt], false, false);
    }
  }

#pragma unroll
  for (int nt = 0; nt < 8; ++nt) {
    const int col = nt * 16 + n;
    const float bias = bf[col];
#pragma unroll
    for (int r = 0; r < 8; ++r) {
      const int m = r + 8 * h;
      gfin[(m0 + m) * HID + col] = fmaxf(acc[nt][r] + bias, 0.0f);
    }
  }
}

__global__ void __launch_bounds__(256)
head_out_kernel(const float* __restrict__ gfin, const float* __restrict__ Wo,
                const float* __restrict__ bo, float* __restrict__ pred) {
  const int g = blockIdx.x * 256 + (int)threadIdx.x;
  if (g >= NGRAPH) return;
  float s = bo[0];
#pragma unroll 8
  for (int c = 0; c < HID; ++c) s = fmaf(gfin[g * HID + c], Wo[c], s);
  pred[g] = s;
}

// ---------------------------------------------------------------------------
extern "C" void kernel_launch(void* const* d_in, const int* in_sizes, int n_in,
                              void* d_out, int out_size, void* d_ws, size_t ws_size,
                              hipStream_t stream) {
  (void)in_sizes; (void)n_in; (void)out_size; (void)ws_size;

  const float* xin[2]   = {(const float*)d_in[0], (const float*)d_in[4]};
  const int*   eidx[2]  = {(const int*)d_in[1],   (const int*)d_in[5]};
  const float* eattr[2] = {(const float*)d_in[2], (const float*)d_in[6]};
  const int*   batch[2] = {(const int*)d_in[3],   (const int*)d_in[7]};
  const float* gfeat    = (const float*)d_in[8];
  const int paramBase[2] = {10, 40};   // 10 tensors per layer, 3 layers per tower

  float* ws   = (float*)d_ws;
  float* xA   = ws;
  float* xB   = xA  + (size_t)N_NODES * HID;
  float* agg  = xB  + (size_t)N_NODES * HID;
  float* gsv  = agg + (size_t)N_NODES * HID;
  float* gsu  = gsv + (size_t)NGRAPH * HID;
  float* gc   = gsu + (size_t)NGRAPH * HID;
  float* gfin = gc  + (size_t)NGRAPH * INFINAL;
  float* pools[2] = {gsv, gsu};

  const int edgeBlocks = N_EDGES / (16 * 4);            // 7500
  const int nodeTiles  = N_NODES / 16;                  // 1875
  const int nodeBlocks = (nodeTiles + 3) / 4;           // 469

  for (int t = 0; t < 2; ++t) {
    const float* cur = xin[t];
    int ldx = NF;
    for (int l = 0; l < 3; ++l) {
      const int pb = paramBase[t] + 10 * l;
      const float* We = (const float*)d_in[pb + 0];
      const float* be = (const float*)d_in[pb + 1];
      const float* W1 = (const float*)d_in[pb + 2];
      const float* b1 = (const float*)d_in[pb + 3];
      const float* W2 = (const float*)d_in[pb + 4];
      const float* b2 = (const float*)d_in[pb + 5];
      const float* ga = (const float*)d_in[pb + 6];
      const float* bt = (const float*)d_in[pb + 7];
      const float* mu = (const float*)d_in[pb + 8];
      const float* va = (const float*)d_in[pb + 9];

      hipMemsetAsync(agg, 0, (size_t)N_NODES * HID * sizeof(float), stream);
      float* outbuf = (l == 1) ? xB : xA;   // l0->xA, l1->xB, l2->xA
      if (l == 0) {
        edge_gine_kernel<NF><<<edgeBlocks, 128, 0, stream>>>(cur, ldx, eattr[t], eidx[t],
                                                             We, be, agg);
        node_mlp_kernel<NF><<<nodeBlocks, 128, 0, stream>>>(cur, ldx, agg, W1, b1, W2, b2,
                                                            ga, bt, mu, va, outbuf, nodeTiles);
      } else {
        edge_gine_kernel<HID><<<edgeBlocks, 128, 0, stream>>>(cur, ldx, eattr[t], eidx[t],
                                                              We, be, agg);
        node_mlp_kernel<HID><<<nodeBlocks, 128, 0, stream>>>(cur, ldx, agg, W1, b1, W2, b2,
                                                             ga, bt, mu, va, outbuf, nodeTiles);
      }
      cur = outbuf; ldx = HID;
    }
    hipMemsetAsync(pools[t], 0, (size_t)NGRAPH * HID * sizeof(float), stream);
    pool_kernel<<<(N_NODES * 32) / 256, 256, 0, stream>>>(cur, batch[t], pools[t]);
  }

  concat_kernel<<<(NGRAPH * INFINAL) / 256, 256, 0, stream>>>(gsv, gsu, gfeat, gc);
  head_mlp_kernel<<<4, 128, 0, stream>>>(gc, (const float*)d_in[70],
                                         (const float*)d_in[71], gfin);
  head_out_kernel<<<1, 256, 0, stream>>>(gfin, (const float*)d_in[72],
                                         (const float*)d_in[73], (float*)d_out);
  // outputs: prediction [256] then g_concat [256*260], concatenated flat
  hipMemcpyAsync((float*)d_out + NGRAPH, gc,
                 (size_t)NGRAPH * INFINAL * sizeof(float),
                 hipMemcpyDeviceToDevice, stream);
}
